// ST_HMR_15599321219550
// MI455X (gfx1250) — matrix operand order repaired and checked by reading, not compiled, not run on hardware
//
#include <hip/hip_runtime.h>

typedef __attribute__((ext_vector_type(16))) __bf16 v16bf;
typedef __attribute__((ext_vector_type(8)))  __bf16 v8bf;
typedef __attribute__((ext_vector_type(8)))  float  v8f;

#define T_STEPS 49
#define NPOS    24
#define BATCH   64
#define HDIM    256
#define GG      9

__device__ __forceinline__ float sigf(float x) { return 1.0f / (1.0f + __expf(-x)); }
__device__ __forceinline__ float tanhfast(float x) {
    float cx = fminf(fmaxf(x, -15.0f), 15.0f);
    float e = __expf(2.0f * cx);
    return (e - 1.0f) / (e + 1.0f);
}

__device__ __forceinline__ v8f wmma_bf16(v16bf a, v16bf b, v8f c) {
    // D = A(16x32 bf16) * B(32x16 bf16) + C(16x16 f32)
    return __builtin_amdgcn_wmma_f32_16x16x32_bf16(false, a, false, b, (short)0, c, false, false);
}

// A-fragment gather from a bf16 row-major [rows][strideK] buffer.
// Layout (16-bit A 16x32, this code's contiguous k order, not the ISA table order): lane m=lane&15 holds row m;
// element e -> K = kbase + 16*(lane>>4) + e  => one contiguous 32B run (two 16B loads).
__device__ __forceinline__ v16bf load_afrag(const __bf16* rowbase_kb, int half) {
    v8bf lo = *(const v8bf*)(rowbase_kb + 16 * half);
    v8bf hi = *(const v8bf*)(rowbase_kb + 16 * half + 8);
    return __builtin_shufflevector(lo, hi, 0,1,2,3,4,5,6,7,8,9,10,11,12,13,14,15);
}

// ---------------------------------------------------------------------------
// zero-init kernel
// ---------------------------------------------------------------------------
__global__ void k_zero(float* p, int n) {
    int i = blockIdx.x * blockDim.x + threadIdx.x;
    int stride = gridDim.x * blockDim.x;
    for (; i < n; i += stride) p[i] = 0.0f;
}

// ---------------------------------------------------------------------------
// per-t weight packing: Ws|Zs (9 gates, K=512, N=256) and Wg|Zg (2x3, K=512, N=256)
// into B-fragment order: [g][coltile16][kb16][lane32][e16] bf16 (32B per lane-frag).
// B layout: col = lane&15, K = kb*32 + 16*(lane>>4) + e.
// ---------------------------------------------------------------------------
__global__ void k_pack(const float* __restrict__ Ws, const float* __restrict__ Zs,
                       const float* __restrict__ Wg, const float* __restrict__ Zg,
                       __bf16* __restrict__ wszs, __bf16* __restrict__ wgzg, int t) {
    const int NWS = GG * 16 * 16 * 32 * 16;   // 1,179,648
    const int NWG = 6  * 16 * 16 * 32 * 16;   //   786,432
    int idx = blockIdx.x * blockDim.x + threadIdx.x;
    int stride = gridDim.x * blockDim.x;
    for (int i = idx; i < NWS + NWG; i += stride) {
        if (i < NWS) {
            int g = i / 131072; int r = i % 131072;
            int ct = r / 8192;  r %= 8192;
            int kb = r / 512;   int r2 = r % 512;
            int lane = r2 / 16, e = r2 % 16;
            int k   = kb * 32 + 16 * (lane >> 4) + e;
            int col = ct * 16 + (lane & 15);
            float v = (k < 256)
                ? Ws[(((long)t * GG + g) * 256 + k) * 256 + col]
                : Zs[(((long)t * GG + g) * 256 + (k - 256)) * 256 + col];
            wszs[i] = (__bf16)v;
        } else {
            int j2 = i - NWS;
            int uj = j2 / 131072; int r = j2 % 131072;
            int u = uj / 3, jj = uj % 3;
            int ct = r / 8192;  r %= 8192;
            int kb = r / 512;   int r2 = r % 512;
            int lane = r2 / 16, e = r2 % 16;
            int k   = kb * 32 + 16 * (lane >> 4) + e;
            int col = ct * 16 + (lane & 15);
            float v = (k < 256)
                ? Wg[((((long)t * 2 + u) * 3 + jj) * 256 + k) * 256 + col]
                : Zg[((((long)t * 2 + u) * 3 + jj) * 256 + (k - 256)) * 256 + col];
            wgzg[j2] = (__bf16)v;
        }
    }
}

// ---------------------------------------------------------------------------
// pre[g][n][b][k] = x_t@U + h_tri@Wt + cm@Zt + bias     (bf16 WMMA, f32 acc)
// grid: (96 row-tiles (n,b16), 2 col-groups of 128, 9 gates), 256 threads.
// A (16 rows x 1280 K) staged fp32->bf16 in LDS; B gathered fp32->bf16 in reg.
// ---------------------------------------------------------------------------
__global__ __launch_bounds__(256)
void k_pre(const float* __restrict__ x, const float* __restrict__ U,
           const float* __restrict__ Wt, const float* __restrict__ Zt,
           const float* __restrict__ bias,
           const float* __restrict__ h_prev, const float* __restrict__ c_prev,
           float* __restrict__ pre, int t) {
    __shared__ __bf16 a_lds[16 * 1280];  // 40 KB
    const int rt = blockIdx.x;           // 0..95
    const int cg = blockIdx.y;           // 0..1
    const int g  = blockIdx.z;           // 0..8
    const int n    = rt >> 2;
    const int brow = (rt & 3) * 16;
    const int tid  = threadIdx.x;

    // Stage A: K regions [0,256)=x_t, [256,1024)=h_tri (n-1,n,n+1), [1024,1280)=cm
    for (int i = tid; i < 16 * 1280; i += 256) {
        int m = i / 1280, k = i % 1280;
        int b = brow + m;
        float v;
        if (k < 256) {
            v = x[(((long)b * T_STEPS + t) * NPOS + n) * HDIM + k];
        } else if (k < 1024) {
            int kk = k - 256; int seg = kk >> 8; int k2 = kk & 255;
            int nn = n + seg - 1;
            v = (nn >= 0 && nn < NPOS) ? h_prev[((long)b * NPOS + nn) * HDIM + k2] : 0.0f;
        } else {
            v = c_prev[((long)b * NPOS + n) * HDIM + (k - 1024)];
        }
        a_lds[i] = (__bf16)v;
    }
    __syncthreads();

    const int wv = tid >> 5, lane = tid & 31;
    const int m = lane & 15, half = lane >> 4;
    const int bcol = cg * 128 + wv * 16 + m;  // output column for B / C / D

    const float* Ub  = U  + (((long)t * GG + g) << 16);
    const float* Wtb = Wt + ((long)t * GG + g) * 768 * 256;
    const float* Ztb = Zt + (((long)t * GG + g) << 16);

    v8f acc = {};
    for (int kb = 0; kb < 40; ++kb) {
        v16bf afrag = load_afrag(&a_lds[m * 1280 + kb * 32], half);
        const float* wp; int krow;
        if (kb < 8)       { wp = Ub;  krow = kb * 32; }
        else if (kb < 32) { wp = Wtb; krow = (kb - 8) * 32; }
        else              { wp = Ztb; krow = (kb - 32) * 32; }
        __builtin_prefetch(wp + (long)(krow + 16 * half + 16) * HDIM + bcol, 0, 1);
        v16bf bfrag;
#pragma unroll
        for (int e = 0; e < 16; ++e) {
            int k = krow + 16 * half + e;
            bfrag[e] = (__bf16)wp[(long)k * HDIM + bcol];
        }
        acc = wmma_bf16(afrag, bfrag, acc);
    }

    float bv = bias[((long)t * GG + g) * HDIM + bcol];
    const int rowb = half * 8;
#pragma unroll
    for (int v = 0; v < 8; ++v) {
        int b = brow + rowb + v;
        pre[((((long)g * NPOS + n) * BATCH + b) << 8) + bcol] = acc[v] + bv;
    }
}

// ---------------------------------------------------------------------------
// joint scan over n=0..23: one 16-wave workgroup per time step.
// LDS: double-buffered [h|c] bf16 A-operand (2 x 64 x 512) + f32 c_joint (64x256).
// Each wave owns 4 fixed 16x16 output tiles across all 9 gates.
// ---------------------------------------------------------------------------
__global__ __launch_bounds__(512)
void k_scan(const float* __restrict__ pre, const __bf16* __restrict__ wszs,
            const float* __restrict__ c_prev,
            const float* __restrict__ gt_cur, const float* __restrict__ gs_cur,
            float* __restrict__ h_next, float* __restrict__ c_next,
            float* __restrict__ h_mean, float* __restrict__ out, int t) {
    extern __shared__ unsigned char smem[];
    __bf16* hc = (__bf16*)smem;                          // 2 x 64 x 512 bf16 = 128 KB
    float*  cj = (float*)(smem + 2 * 64 * 512 * 2);      // 64 x 256 f32     =  64 KB

    const int tid = threadIdx.x;
    for (int i = tid; i < 2 * 64 * 512; i += 512) hc[i] = (__bf16)0.0f;
    for (int i = tid; i < 64 * 256;     i += 512) cj[i] = 0.0f;
    __syncthreads();

    const int w = tid >> 5, lane = tid & 31;
    const int m = lane & 15, half = lane >> 4;

    float hsum[4][8];
#pragma unroll
    for (int i = 0; i < 4; ++i)
#pragma unroll
        for (int v = 0; v < 8; ++v) hsum[i][v] = 0.0f;

    for (int n = 0; n < NPOS; ++n) {
        const __bf16* hcr = hc + ((n)     & 1) * 64 * 512;
        __bf16*       hcw = hc + ((n + 1) & 1) * 64 * 512;

        for (int i = 0; i < 4; ++i) {
            const int tileId = w * 4 + i;
            const int tb = tileId >> 4, tc = tileId & 15;
            const int col = tc * 16 + m;

            v8f acc[9];
#pragma unroll
            for (int g = 0; g < 9; ++g) {
#pragma unroll
                for (int v = 0; v < 8; ++v) {
                    int b = tb * 16 + half * 8 + v;
                    acc[g][v] = pre[((((long)g * NPOS + n) * BATCH + b) << 8) + col];
                }
            }

            for (int kb = 0; kb < 16; ++kb) {
                v16bf afrag = load_afrag(&hcr[(tb * 16 + m) * 512 + kb * 32], half);
#pragma unroll
                for (int g = 0; g < 9; ++g) {
                    v16bf bfrag = *((const v16bf*)wszs + (((g * 16 + tc) * 16 + kb) * 32 + lane));
                    acc[g] = wmma_bf16(afrag, bfrag, acc[g]);
                }
            }

#pragma unroll
            for (int v = 0; v < 8; ++v) {
                int b = tb * 16 + half * 8 + v;
                float iG  = sigf(acc[0][v]);
                float lt  = sigf(acc[1][v]);
                float ft  = sigf(acc[2][v]);
                float rtg = sigf(acc[3][v]);
                float sg  = sigf(acc[4][v]);
                float gtg = sigf(acc[5][v]);
                float gsg = sigf(acc[6][v]);
                float o   = sigf(acc[7][v]);
                float cand = tanhfast(acc[8][v]);

                float cl = (n > 0)        ? c_prev[((long)b * NPOS + (n - 1)) * HDIM + col] : 0.0f;
                float cm = c_prev[((long)b * NPOS + n) * HDIM + col];
                float cr = (n < NPOS - 1) ? c_prev[((long)b * NPOS + (n + 1)) * HDIM + col] : 0.0f;
                float cs  = cj[b * 256 + col];
                float gtv = gt_cur[b * 256 + col];
                float gsv = gs_cur[b * 256 + col];

                float cn = lt * cl + ft * cm + rtg * cr + sg * cs + gtg * gtv + gsg * gsv + iG * cand;
                float hn = o * tanhfast(cn);

                cj[b * 256 + col]        = cn;
                hcw[b * 512 + col]       = (__bf16)hn;
                hcw[b * 512 + 256 + col] = (__bf16)cn;
                h_next[((long)b * NPOS + n) * HDIM + col] = hn;
                c_next[((long)b * NPOS + n) * HDIM + col] = cn;
                out[(((long)b * T_STEPS + t) * NPOS + n) * HDIM + col] = hn;
                hsum[i][v] += hn;
            }
        }
        __syncthreads();
    }

#pragma unroll
    for (int i = 0; i < 4; ++i) {
        const int tileId = w * 4 + i;
        const int tb = tileId >> 4, tc = tileId & 15;
        const int col = tc * 16 + m;
#pragma unroll
        for (int v = 0; v < 8; ++v) {
            int b = tb * 16 + half * 8 + v;
            h_mean[b * 256 + col] = hsum[i][v] * (1.0f / NPOS);
        }
    }
}

// ---------------------------------------------------------------------------
// gate-state update: g_new = o * tanh(f*g + (1-f)*cg); 2 blocks (u=t/s), 16 waves.
// ---------------------------------------------------------------------------
__global__ __launch_bounds__(512)
void k_gupd(const float* __restrict__ h_mean, const __bf16* __restrict__ wgzg,
            const float* __restrict__ bg,
            const float* __restrict__ gt_cur, const float* __restrict__ gs_cur,
            float* __restrict__ gt_next, float* __restrict__ gs_next, int t) {
    extern __shared__ unsigned char smem[];
    __bf16* ag = (__bf16*)smem;  // 64 x 512 bf16 = 64 KB
    const int u = blockIdx.x;
    const float* gc = u ? gs_cur : gt_cur;
    float*       gn = u ? gs_next : gt_next;
    const int tid = threadIdx.x;

    for (int i = tid; i < 64 * 512; i += 512) {
        int b = i >> 9, k = i & 511;
        float v = (k < 256) ? h_mean[b * 256 + k] : gc[b * 256 + (k - 256)];
        ag[i] = (__bf16)v;
    }
    __syncthreads();

    const int w = tid >> 5, lane = tid & 31;
    const int m = lane & 15, half = lane >> 4;

    for (int i = 0; i < 4; ++i) {
        const int tileId = w * 4 + i;
        const int tb = tileId >> 4, tc = tileId & 15;
        const int col = tc * 16 + m;

        v8f acc[3];
#pragma unroll
        for (int j = 0; j < 3; ++j) {
            float bv = bg[(((long)t * 2 + u) * 3 + j) * HDIM + col];
#pragma unroll
            for (int v = 0; v < 8; ++v) acc[j][v] = bv;
        }

        for (int kb = 0; kb < 16; ++kb) {
            v16bf afrag = load_afrag(&ag[(tb * 16 + m) * 512 + kb * 32], half);
#pragma unroll
            for (int j = 0; j < 3; ++j) {
                v16bf bfrag = *((const v16bf*)wgzg + ((((u * 3 + j) * 16 + tc) * 16 + kb) * 32 + lane));
                acc[j] = wmma_bf16(afrag, bfrag, acc[j]);
            }
        }

#pragma unroll
        for (int v = 0; v < 8; ++v) {
            int b = tb * 16 + half * 8 + v;
            float f  = sigf(acc[0][v]);
            float cg = tanhfast(acc[1][v]);
            float o  = sigf(acc[2][v]);
            float gv = gc[b * 256 + col];
            gn[b * 256 + col] = o * tanhfast(f * gv + (1.0f - f) * cg);
        }
    }
}

// ---------------------------------------------------------------------------
extern "C" void kernel_launch(void* const* d_in, const int* in_sizes, int n_in,
                              void* d_out, int out_size, void* d_ws, size_t ws_size,
                              hipStream_t stream) {
    const float* x  = (const float*)d_in[0];
    const float* U  = (const float*)d_in[1];
    const float* Wt = (const float*)d_in[2];
    const float* Ws = (const float*)d_in[3];
    const float* Zt = (const float*)d_in[4];
    const float* Zs = (const float*)d_in[5];
    const float* bb = (const float*)d_in[6];
    const float* Wg = (const float*)d_in[7];
    const float* Zg = (const float*)d_in[8];
    const float* bg = (const float*)d_in[9];
    float* out = (float*)d_out;

    unsigned char* ws = (unsigned char*)d_ws;
    size_t off = 0;
    auto alloc = [&](size_t bytes) -> void* {
        void* p = ws + off;
        off += (bytes + 255) & ~(size_t)255;
        return p;
    };
    const size_t SZ_STATE = (size_t)BATCH * NPOS * HDIM * sizeof(float);   // 1.5 MB
    const size_t SZ_G     = (size_t)BATCH * HDIM * sizeof(float);          // 64 KB
    const size_t SZ_PRE   = (size_t)GG * NPOS * BATCH * HDIM * sizeof(float);
    const size_t SZ_WSZS  = (size_t)GG * 512 * 256 * sizeof(__bf16);
    const size_t SZ_WGZG  = (size_t)6  * 512 * 256 * sizeof(__bf16);

    float*  h_state[2] = { (float*)alloc(SZ_STATE), (float*)alloc(SZ_STATE) };
    float*  c_state[2] = { (float*)alloc(SZ_STATE), (float*)alloc(SZ_STATE) };
    float*  g_t[2]     = { (float*)alloc(SZ_G), (float*)alloc(SZ_G) };
    float*  g_s[2]     = { (float*)alloc(SZ_G), (float*)alloc(SZ_G) };
    float*  h_mean     = (float*)alloc(SZ_G);
    float*  pre        = (float*)alloc(SZ_PRE);
    __bf16* wszs       = (__bf16*)alloc(SZ_WSZS);
    __bf16* wgzg       = (__bf16*)alloc(SZ_WGZG);

    // deterministic start: zero t=0 parity state every call
    k_zero<<<1024, 256, 0, stream>>>(h_state[0], BATCH * NPOS * HDIM);
    k_zero<<<1024, 256, 0, stream>>>(c_state[0], BATCH * NPOS * HDIM);
    k_zero<<<64,   256, 0, stream>>>(g_t[0], BATCH * HDIM);
    k_zero<<<64,   256, 0, stream>>>(g_s[0], BATCH * HDIM);

    for (int t = 0; t < T_STEPS; ++t) {
        const int p = t & 1;
        k_pack<<<2048, 256, 0, stream>>>(Ws, Zs, Wg, Zg, wszs, wgzg, t);
        k_pre<<<dim3(96, 2, 9), 256, 0, stream>>>(x, U, Wt, Zt, bb,
                                                  h_state[p], c_state[p], pre, t);
        k_scan<<<1, 512, 196608, stream>>>(pre, wszs, c_state[p], g_t[p], g_s[p],
                                           h_state[1 - p], c_state[1 - p],
                                           h_mean, out, t);
        k_gupd<<<2, 512, 65536, stream>>>(h_mean, wgzg, bg, g_t[p], g_s[p],
                                          g_t[1 - p], g_s[1 - p], t);
    }
    (void)in_sizes; (void)n_in; (void)out_size; (void)ws_size;
}
